// MultiHeadedAttention_10926396801348
// MI455X (gfx1250) — matrix operand order repaired and checked by reading, not compiled, not run on hardware
//
#include <hip/hip_runtime.h>
#include <hip/hip_bf16.h>

// ---------------- types ----------------
typedef __bf16 bf16;
typedef __attribute__((ext_vector_type(16))) __bf16 v16bf;
typedef __attribute__((ext_vector_type(8)))  float  v8f;

static __device__ __forceinline__ v8f wmma_bf16(v16bf a, v16bf b, v8f c) {
  // D = A(16x32 bf16) * B(32x16 bf16) + C(16x16 f32)
  return __builtin_amdgcn_wmma_f32_16x16x32_bf16(false, a, false, b, (short)0, c,
                                                 false, false);
}

// A-fragment: contiguous k order used here (B matches) = one 16-elem run per lane:
// k = kb + half*16 + [0..15]
static __device__ __forceinline__ v16bf load_afrag(const bf16* rowk, int half) {
  v16bf a;
  const bf16* p = rowk + half * 16;
#pragma unroll
  for (int e = 0; e < 16; ++e) { a[e] = p[e]; }
  return a;
}

// Problem constants (B=2, T=8, C=256, H=W=64)
#define BT   16
#define TT   8
#define CC   256
#define HH   64
#define WW   64
#define PP   4096                 // H*W
#define SLAB 4194304u             // per-scale windowed slab (elems)
#define TEN  16777216u            // full tensor elems: BT*C*H*W

// ======================================================================
// Pre-pass A: weights -> bf16.  wo/wd repacked [tap][oc][ic].
// ======================================================================
__global__ __launch_bounds__(256) void k_pack_w(
    const float* __restrict__ wq, const float* __restrict__ wk,
    const float* __restrict__ wv, const float* __restrict__ wo,
    const float* __restrict__ wd, bf16* __restrict__ wqb,
    bf16* __restrict__ wkb, bf16* __restrict__ wvb, bf16* __restrict__ wob,
    bf16* __restrict__ wdb) {
  const int i = blockIdx.x * 256 + threadIdx.x;
  if (i < 65536) {
    wqb[i] = (bf16)wq[i];
    wkb[i] = (bf16)wk[i];
    wvb[i] = (bf16)wv[i];
  }
  if (i < 589824) {                       // 9 * 256 * 256
    const int t9 = i >> 16, rem = i & 65535;
    const int oc = rem >> 8, ic = rem & 255;
    const size_t src = (size_t)(oc * 256 + ic) * 9 + t9;
    wob[i] = (bf16)wo[src];
    wdb[i] = (bf16)wd[src];
  }
}

// ======================================================================
// Pre-pass B: x NCHW f32 -> NHWC bf16 (LDS-tiled 32x32 transpose).
// ======================================================================
__global__ __launch_bounds__(256) void k_xpose(const float* __restrict__ x,
                                               bf16* __restrict__ xb) {
  __shared__ float t[32][33];
  const int n = blockIdx.z;
  const int pb = blockIdx.x * 32, cb = blockIdx.y * 32;
  const int tx = threadIdx.x & 31, ty = threadIdx.x >> 5;   // 32 x 8
  const float* xn = x + (size_t)n * CC * PP;
#pragma unroll
  for (int i = 0; i < 32; i += 8)
    t[ty + i][tx] = xn[(size_t)(cb + ty + i) * PP + pb + tx];
  __syncthreads();
  bf16* out = xb + (size_t)n * PP * CC;
#pragma unroll
  for (int i = 0; i < 32; i += 8)
    out[(size_t)(pb + ty + i) * CC + cb + tx] = (bf16)t[tx][ty + i];
}

// ======================================================================
// Kernel 1: Q/K/V 1x1-conv GEMM + depthwise 3x3 into V.
// B operand: x NHWC bf16 (contiguous fragments).  Q/K packed windowed
// [scale][b][L][D]; V packed TRANSPOSED [scale][b][D][L].
// ======================================================================
__global__ __launch_bounds__(128) void k_qkv(
    const float* __restrict__ x, const bf16* __restrict__ xb,
    const bf16* __restrict__ wqb, const float* __restrict__ bq,
    const bf16* __restrict__ wkb, const float* __restrict__ bk,
    const bf16* __restrict__ wvb, const float* __restrict__ bv,
    const float* __restrict__ wle, const float* __restrict__ ble,
    bf16* __restrict__ QW, bf16* __restrict__ KW, bf16* __restrict__ VWT) {
  const int tid = threadIdx.x, lane = tid & 31, w = tid >> 5;
  const int n16 = lane & 15, half = lane >> 4;
  const int p0  = blockIdx.x * 16;
  const int oc0 = (blockIdx.y * 4 + w) * 16;
  const int n   = blockIdx.z;

  const bf16* brow = xb + ((size_t)n * PP + p0 + n16) * CC;   // lane's column
  const bf16* qr = wqb + (size_t)(oc0 + n16) * CC;            // lane's A row
  const bf16* kr = wkb + (size_t)(oc0 + n16) * CC;
  const bf16* vr = wvb + (size_t)(oc0 + n16) * CC;

  v8f cq = {}, ck = {}, cv = {};
#pragma unroll
  for (int k0 = 0; k0 < CC; k0 += 32) {
    v16bf bx = *(const v16bf*)(brow + k0 + half * 16);
    cq = wmma_bf16(load_afrag(qr + k0, half), bx, cq);
    ck = wmma_bf16(load_afrag(kr + k0, half), bx, ck);
    cv = wmma_bf16(load_afrag(vr + k0, half), bx, cv);
  }

  // epilogue: bias, depthwise-3x3 into V, pack into windowed layout
  const int t = n & 7, bb = n >> 3;
  const float* xn = x + (size_t)n * CC * PP;
#pragma unroll
  for (int r = 0; r < 8; ++r) {
    const int oc = oc0 + r + 8 * half;             // C-frag row
    const int p  = p0 + n16;                       // C-frag col
    const int y = p >> 6, xc = p & 63;
    float vq = cq[r] + bq[oc];
    float vk = ck[r] + bk[oc];
    float vv = cv[r] + bv[oc] + ble[oc];
    const float* wl  = wle + oc * 9;
    const float* xch = xn + (size_t)oc * PP;
#pragma unroll
    for (int dy = 0; dy < 3; ++dy)
#pragma unroll
      for (int dx = 0; dx < 3; ++dx) {
        int yy = y + dy - 1, xx = xc + dx - 1;
        if (yy >= 0 && yy < HH && xx >= 0 && xx < WW)
          vv += wl[dy * 3 + dx] * xch[yy * WW + xx];
      }
    const int s  = oc >> 6, c = oc & 63;
    const int pw = 16 >> s;                        // patch size 16,8,4,2
    const int outW = WW / pw;
    const int l  = (t * outW + (y / pw)) * outW + (xc / pw);
    const int d  = (c * pw + (y % pw)) * pw + (xc % pw);
    const int Ls = 128 << (2 * s);
    const int Ds = 16384 >> (2 * s);
    const size_t sb = (size_t)s * SLAB;
    QW[sb + ((size_t)bb * Ls + l) * Ds + d]  = (bf16)vq;
    KW[sb + ((size_t)bb * Ls + l) * Ds + d]  = (bf16)vk;
    VWT[sb + ((size_t)bb * Ds + d) * Ls + l] = (bf16)vv;     // transposed
  }
}

// ======================================================================
// Kernel 2: attention for scales with L<=2048 (patch 16/8/4).
// Phase1: S[16,L] -> LDS bf16, softmax, Phase2: P@V (V transposed),
// scatter to YW in NHWC bf16.
// ======================================================================
__global__ __launch_bounds__(128) void k_attn_small(
    const bf16* __restrict__ QW, const bf16* __restrict__ KW,
    const bf16* __restrict__ VWT, bf16* __restrict__ YW,
    int s, int L, int D, int patch) {
  __shared__ bf16 S[16 * 2048];                    // 64 KB max (L=2048)
  const int tid = threadIdx.x, lane = tid & 31, w = tid >> 5;
  const int n16 = lane & 15, half = lane >> 4;
  const int nq = L >> 4;
  const int bb = blockIdx.x / nq, qt = blockIdx.x % nq;
  const size_t base = (size_t)s * SLAB + (size_t)bb * ((size_t)L * D);
  const bf16* Q  = QW + base;
  const bf16* K  = KW + base;
  const bf16* VT = VWT + base;                     // [D][L]
  const float scale = rsqrtf((float)D);
  const int q0 = qt * 16;

  // ---- phase 1: scores ----
  const bf16* qrow = Q + (size_t)(q0 + n16) * D;
  for (int mt = w; mt < nq; mt += 4) {
    const int m0 = mt * 16;
    const bf16* krow = K + (size_t)(m0 + n16) * D;
    v8f acc = {};
    for (int kb = 0; kb < D; kb += 32) {
      v16bf a = load_afrag(qrow + kb, half);
      v16bf b = *(const v16bf*)(krow + kb + half * 16);
      acc = wmma_bf16(a, b, acc);
    }
#pragma unroll
    for (int r = 0; r < 8; ++r)
      S[(r + 8 * half) * L + m0 + n16] = (bf16)(acc[r] * scale);
  }
  __syncthreads();

  // ---- softmax: 8 threads per row ----
  {
    const int row = tid >> 3, sub = tid & 7;
    float mx = -3.0e38f;
    for (int j = sub; j < L; j += 8) mx = fmaxf(mx, (float)S[row * L + j]);
    mx = fmaxf(mx, __shfl_xor(mx, 1, 32));
    mx = fmaxf(mx, __shfl_xor(mx, 2, 32));
    mx = fmaxf(mx, __shfl_xor(mx, 4, 32));
    float sum = 0.f;
    for (int j = sub; j < L; j += 8) sum += __expf((float)S[row * L + j] - mx);
    sum += __shfl_xor(sum, 1, 32);
    sum += __shfl_xor(sum, 2, 32);
    sum += __shfl_xor(sum, 4, 32);
    const float inv = 1.f / sum;
    for (int j = sub; j < L; j += 8)
      S[row * L + j] = (bf16)(__expf((float)S[row * L + j] - mx) * inv);
  }
  __syncthreads();

  // ---- phase 2: y = P @ V, scatter NHWC ----
  const int ndt = D >> 4;
  const int outW = WW / patch;
  for (int dt = w; dt < ndt; dt += 4) {
    const int d0 = dt * 16;
    const bf16* vrow = VT + (size_t)(d0 + n16) * L;            // lane's column
    v8f acc = {};
    for (int kb = 0; kb < L; kb += 32) {
      v16bf a = load_afrag(&S[n16 * L + kb], half);
      v16bf b = *(const v16bf*)(vrow + kb + half * 16);
      acc = wmma_bf16(a, b, acc);
    }
    const int d = d0 + n16;
    const int c = d / (patch * patch);
    const int r2 = d % (patch * patch);
    const int py = r2 / patch, px = r2 % patch;
#pragma unroll
    for (int r = 0; r < 8; ++r) {
      const int l = q0 + r + 8 * half;
      const int t = l / (outW * outW);
      const int rem = l % (outW * outW);
      const int oh = rem / outW, ow = rem % outW;
      const int nimg = bb * TT + t, ch = s * 64 + c;
      const int yy = oh * patch + py, xx = ow * patch + px;
      YW[((size_t)nimg * PP + yy * WW + xx) * CC + ch] = (bf16)acc[r];
    }
  }
}

// ======================================================================
// Kernel 3: flash attention for scale 3 (patch 2, L=8192, D=256).
// ======================================================================
__global__ __launch_bounds__(128) void k_attn_flash(
    const bf16* __restrict__ QW, const bf16* __restrict__ KW,
    const bf16* __restrict__ VWT, bf16* __restrict__ YW) {
  const int L = 8192, D = 256;
  __shared__ bf16 P[4][16][34];                    // per-wave transpose tile
  const int tid = threadIdx.x, lane = tid & 31, w = tid >> 5;
  const int n16 = lane & 15, half = lane >> 4;
  const int gq = blockIdx.x * 4 + w;               // 0..1023
  const int bb = gq >> 9, qt = gq & 511, q0 = qt * 16;
  const size_t base = (size_t)3 * SLAB + (size_t)bb * ((size_t)L * D);
  const bf16* Q  = QW + base;
  const bf16* K  = KW + base;
  const bf16* VT = VWT + base;                     // [D][L]
  const float scale = 0.0625f;                     // 1/sqrt(256)

  v8f o[16];
#pragma unroll
  for (int dt = 0; dt < 16; ++dt) o[dt] = (v8f){};
  float rmax[8], rsum[8];
#pragma unroll
  for (int r = 0; r < 8; ++r) { rmax[r] = -3.0e38f; rsum[r] = 0.f; }

  const bf16* qrow = Q + (size_t)(q0 + n16) * D;
  for (int m0 = 0; m0 < L; m0 += 32) {
    // ---- S tiles for columns m0..m0+31 ----
    v8f s0 = {}, s1 = {};
    const bf16* k0r = K + (size_t)(m0 + n16) * D;
    const bf16* k1r = K + (size_t)(m0 + 16 + n16) * D;
#pragma unroll
    for (int kb = 0; kb < 256; kb += 32) {
      v16bf a = load_afrag(qrow + kb, half);
      v16bf b0 = *(const v16bf*)(k0r + kb + half * 16);
      v16bf b1 = *(const v16bf*)(k1r + kb + half * 16);
      s0 = wmma_bf16(a, b0, s0);
      s1 = wmma_bf16(a, b1, s1);
    }
    // ---- online softmax update (per row, 16-lane halves) ----
    __syncthreads();
#pragma unroll
    for (int r = 0; r < 8; ++r) {
      float v0 = s0[r] * scale, v1 = s1[r] * scale;
      float mx = fmaxf(v0, v1);
      mx = fmaxf(mx, __shfl_xor(mx, 1, 32));
      mx = fmaxf(mx, __shfl_xor(mx, 2, 32));
      mx = fmaxf(mx, __shfl_xor(mx, 4, 32));
      mx = fmaxf(mx, __shfl_xor(mx, 8, 32));
      const float Mn = fmaxf(rmax[r], mx);
      const float alpha = __expf(rmax[r] - Mn);
      const float p0 = __expf(v0 - Mn), p1 = __expf(v1 - Mn);
      float rs = p0 + p1;
      rs += __shfl_xor(rs, 1, 32);
      rs += __shfl_xor(rs, 2, 32);
      rs += __shfl_xor(rs, 4, 32);
      rs += __shfl_xor(rs, 8, 32);
      rsum[r] = rsum[r] * alpha + rs;
      rmax[r] = Mn;
#pragma unroll
      for (int dt = 0; dt < 16; ++dt) o[dt][r] *= alpha;
      P[w][r + 8 * half][n16]      = (bf16)p0;
      P[w][r + 8 * half][16 + n16] = (bf16)p1;
    }
    __syncthreads();
    // ---- P @ V (V transposed: contiguous B fragments) ----
    v16bf ap = load_afrag(&P[w][n16][0], half);
#pragma unroll
    for (int dt = 0; dt < 16; ++dt) {
      v16bf b =
          *(const v16bf*)(VT + (size_t)(dt * 16 + n16) * L + m0 + half * 16);
      o[dt] = wmma_bf16(ap, b, o[dt]);
    }
  }

  // ---- normalize + scatter NHWC (patch=2, outW=32, ch 192..255) ----
  float inv[8];
#pragma unroll
  for (int r = 0; r < 8; ++r) inv[r] = 1.f / rsum[r];
#pragma unroll
  for (int dt = 0; dt < 16; ++dt) {
    const int d = dt * 16 + n16;
    const int c = d >> 2, r2 = d & 3, py = r2 >> 1, px = r2 & 1;
#pragma unroll
    for (int r = 0; r < 8; ++r) {
      const int l = q0 + r + 8 * half;
      const int t = l >> 10, rem = l & 1023;
      const int oh = rem >> 5, ow = rem & 31;
      const int nimg = bb * TT + t, ch = 192 + c;
      const int yy = oh * 2 + py, xx = ow * 2 + px;
      YW[((size_t)nimg * PP + yy * WW + xx) * CC + ch] =
          (bf16)(o[dt][r] * inv[r]);
    }
  }
}

// ======================================================================
// Kernel 4: both 3x3 convs as 9 shifted 1x1 GEMMs over NHWC input.
// B fragment = contiguous 32B channel run (or zero at padding).
// ======================================================================
__global__ __launch_bounds__(128) void k_outconv(
    const bf16* __restrict__ YW, const bf16* __restrict__ wob,
    const float* __restrict__ bo, const bf16* __restrict__ wdb,
    const float* __restrict__ bd, float* __restrict__ xout,
    float* __restrict__ dout) {
  const int tid = threadIdx.x, lane = tid & 31, w = tid >> 5;
  const int n16 = lane & 15, half = lane >> 4;
  const int p0  = blockIdx.x * 16;
  const int oc0 = (blockIdx.y * 4 + w) * 16;
  const int n   = blockIdx.z;
  const bf16* in = YW + (size_t)n * PP * CC;       // NHWC

  const int p = p0 + n16, y = p >> 6, xc = p & 63;

  v8f co = {}, cd = {};
#pragma unroll
  for (int t9 = 0; t9 < 9; ++t9) {
    const int dy = t9 / 3, dx = t9 - dy * 3;
    const int yy = y + dy - 1, xx = xc + dx - 1;
    const bool inb = (yy >= 0 && yy < HH && xx >= 0 && xx < WW);
    const bf16* bcol = in + ((size_t)(yy * WW + xx)) * CC;
    const bf16* ar0 = wob + ((size_t)t9 * 65536) + (size_t)(oc0 + n16) * CC;
    const bf16* ar1 = wdb + ((size_t)t9 * 65536) + (size_t)(oc0 + n16) * CC;
#pragma unroll
    for (int k0 = 0; k0 < CC; k0 += 32) {
      v16bf b = {};
      if (inb) b = *(const v16bf*)(bcol + k0 + half * 16);
      co = wmma_bf16(load_afrag(ar0 + k0, half), b, co);
      cd = wmma_bf16(load_afrag(ar1 + k0, half), b, cd);
    }
  }
#pragma unroll
  for (int r = 0; r < 8; ++r) {
    const int oc = oc0 + r + 8 * half;
    float vo = co[r] + bo[oc];
    float vd = cd[r] + bd[oc];
    vo = vo > 0.f ? vo : 0.2f * vo;
    vd = vd > 0.f ? vd : 0.2f * vd;
    const size_t idx = ((size_t)n * CC + oc) * PP + p;   // NCHW f32 outputs
    xout[idx] = vo;
    dout[idx] = vd;
  }
}

// ======================================================================
extern "C" void kernel_launch(void* const* d_in, const int* in_sizes, int n_in,
                              void* d_out, int out_size, void* d_ws,
                              size_t ws_size, hipStream_t stream) {
  const float* x   = (const float*)d_in[0];
  // d_in[1]=m (no-op mask in reference), d_in[2]=depth_map (unused)
  const float* wq  = (const float*)d_in[3];
  const float* bq  = (const float*)d_in[4];
  const float* wk  = (const float*)d_in[5];
  const float* bk  = (const float*)d_in[6];
  const float* wv  = (const float*)d_in[7];
  const float* bv  = (const float*)d_in[8];
  const float* wle = (const float*)d_in[9];
  const float* ble = (const float*)d_in[10];
  const float* wo  = (const float*)d_in[11];
  const float* bo  = (const float*)d_in[12];
  const float* wd  = (const float*)d_in[13];
  const float* bd  = (const float*)d_in[14];

  bf16* QW  = (bf16*)d_ws;         // 32 MB each
  bf16* KW  = QW + TEN;
  bf16* VWT = KW + TEN;            // V transposed [scale][b][D][L]
  bf16* YW  = VWT + TEN;           // attention output, NHWC bf16
  bf16* XB  = YW + TEN;            // x in NHWC bf16
  bf16* WQB = XB + TEN;
  bf16* WKB = WQB + 65536;
  bf16* WVB = WKB + 65536;
  bf16* WOB = WVB + 65536;         // [9][oc][ic]
  bf16* WDB = WOB + 589824;

  float* xout = (float*)d_out;
  float* dout = xout + TEN;

  // 0) pre-pass packing
  k_pack_w<<<dim3(2304), 256, 0, stream>>>(wq, wk, wv, wo, wd, WQB, WKB, WVB,
                                           WOB, WDB);
  k_xpose<<<dim3(128, 8, 16), 256, 0, stream>>>(x, XB);
  // 1) QKV projection + depthwise, packed windowed (V transposed)
  k_qkv<<<dim3(256, 4, 16), 128, 0, stream>>>(x, XB, WQB, bq, WKB, bk, WVB, bv,
                                              wle, ble, QW, KW, VWT);
  // 2) attention, scales 0..2 (patch 16, 8, 4)
  k_attn_small<<<dim3(16), 128, 0, stream>>>(QW, KW, VWT, YW, 0, 128, 16384, 16);
  k_attn_small<<<dim3(64), 128, 0, stream>>>(QW, KW, VWT, YW, 1, 512, 4096, 8);
  k_attn_small<<<dim3(256), 128, 0, stream>>>(QW, KW, VWT, YW, 2, 2048, 1024, 4);
  // 3) flash attention, scale 3 (patch 2)
  k_attn_flash<<<dim3(256), 128, 0, stream>>>(QW, KW, VWT, YW);
  // 4) output convs + leaky relu
  k_outconv<<<dim3(256, 4, 16), 128, 0, stream>>>(YW, WOB, bo, WDB, bd, xout,
                                                  dout);
}
